// MambaBlock_10479720203016
// MI455X (gfx1250) — compile-verified, matrix operand order checked by reading, not run on hardware
//
#include <hip/hip_runtime.h>
#include <cstdint>
#include <cstddef>

// ---------------------------------------------------------------------------
// Mamba block for MI455X (gfx1250): bf16 WMMA GEMMs + LDS-broadcast scan.
// ---------------------------------------------------------------------------

typedef __attribute__((ext_vector_type(16))) __bf16 v16bf;
typedef __attribute__((ext_vector_type(8)))  __bf16 v8bf;
typedef __attribute__((ext_vector_type(8)))  float  v8f;

#define D_MODEL 1024
#define D_INNER 2048
#define D_STATE 16
#define D_CONV  4
#define DT_RANK 64
#define BATCH   4
#define SEQ     4096
#define MROWS   (BATCH * SEQ)            /* 16384 */
#define NPROJ   (DT_RANK + 2 * D_STATE)  /* 96    */

static __device__ __forceinline__ float silu_f(float v) {
  return v / (1.0f + __expf(-v));
}

// ---------------- elementwise fp32 -> bf16 ----------------
__global__ void k_f32_to_bf16(const float* __restrict__ src,
                              __bf16* __restrict__ dst, size_t n) {
  size_t i = (size_t)blockIdx.x * blockDim.x + threadIdx.x;
  if (i < n) dst[i] = (__bf16)src[i];
}

// ---------------- generic bf16 WMMA GEMM: C = A(MxK) * W(NxK)^T ------------
// Block: 256 threads = 8 wave32. Block tile 128(M) x 128(N).
// Wave tile 64(M) x 32(N) = 4x2 WMMA 16x16 tiles; K stepped by 32.
__global__ __launch_bounds__(256)
void k_gemm_bf16_wmma(const __bf16* __restrict__ A, const __bf16* __restrict__ W,
                      float* __restrict__ Cf, __bf16* __restrict__ Cb,
                      int M, int N, int K) {
  const int lane = threadIdx.x & 31;
  const int wave = threadIdx.x >> 5;
  const int wm = wave & 1;   // 2 waves along M
  const int wn = wave >> 1;  // 4 waves along N
  const int m0 = blockIdx.y * 128 + wm * 64;
  const int n0 = blockIdx.x * 128 + wn * 32;

  v8f acc[4][2];
#pragma unroll
  for (int i = 0; i < 4; ++i)
#pragma unroll
    for (int j = 0; j < 2; ++j)
#pragma unroll
      for (int e = 0; e < 8; ++e) acc[i][j][e] = 0.0f;

  // ISA 16-bit A/B fragment layout (wave32):
  //   lanes 0-15  : row = lane,      K in {k0+0..7,  k0+16..23}
  //   lanes 16-31 : row = lane - 16, K in {k0+8..15, k0+24..31}
  const int rsel = lane & 15;
  const int ksel = (lane >> 4) << 3;

  for (int k0 = 0; k0 < K; k0 += 32) {
    v16bf af[4], bfr[2];
#pragma unroll
    for (int i = 0; i < 4; ++i) {
      int r = m0 + i * 16 + rsel;
      r = (r < M) ? r : (M - 1);
      const __bf16* p = A + (size_t)r * K + (k0 + ksel);
      v8bf lo = *(const v8bf*)p;
      v8bf hi = *(const v8bf*)(p + 16);
#pragma unroll
      for (int e = 0; e < 8; ++e) { af[i][e] = lo[e]; af[i][e + 8] = hi[e]; }
    }
#pragma unroll
    for (int j = 0; j < 2; ++j) {
      int r = n0 + j * 16 + rsel;
      r = (r < N) ? r : (N - 1);
      const __bf16* p = W + (size_t)r * K + (k0 + ksel);
      v8bf lo = *(const v8bf*)p;
      v8bf hi = *(const v8bf*)(p + 16);
#pragma unroll
      for (int e = 0; e < 8; ++e) { bfr[j][e] = lo[e]; bfr[j][e + 8] = hi[e]; }
    }
#pragma unroll
    for (int i = 0; i < 4; ++i)
#pragma unroll
      for (int j = 0; j < 2; ++j)
        acc[i][j] = __builtin_amdgcn_wmma_f32_16x16x32_bf16(
            false, af[i], false, bfr[j], (short)0, acc[i][j], false, false);
  }

  // C/D layout: lanes 0-15 -> rows m0..m0+7 (VGPR v = row m0+v), col = lane;
  //             lanes 16-31 -> rows m0+8..m0+15, col = lane-16.
  const int colsel = lane & 15;
  const int rowsel = (lane >> 4) << 3;
#pragma unroll
  for (int i = 0; i < 4; ++i) {
#pragma unroll
    for (int j = 0; j < 2; ++j) {
      int col = n0 + j * 16 + colsel;
      int row = m0 + i * 16 + rowsel;
      if (col < N && (row + 7) < M) {
        if (Cb) {
#pragma unroll
          for (int v = 0; v < 8; ++v)
            Cb[(size_t)(row + v) * N + col] = (__bf16)acc[i][j][v];
        } else {
#pragma unroll
          for (int v = 0; v < 8; ++v)
            Cf[(size_t)(row + v) * N + col] = acc[i][j][v];
        }
      }
    }
  }
}

// ---------------- causal depthwise conv1d + SiLU ----------------
// xz is (B*T, 2*D_INNER) bf16; x_inner occupies columns [0, D_INNER).
__global__ void k_conv_silu(const __bf16* __restrict__ xz,
                            const float* __restrict__ cw,
                            const float* __restrict__ cb,
                            __bf16* __restrict__ xconv) {
  size_t i = (size_t)blockIdx.x * blockDim.x + threadIdx.x;
  if (i >= (size_t)MROWS * D_INNER) return;
  int d = (int)(i % D_INNER);
  size_t bt = i / D_INNER;
  int t = (int)(bt % SEQ);
  float acc = cb[d];
#pragma unroll
  for (int j = 0; j < D_CONV; ++j) {
    int lag = D_CONV - 1 - j;  // tap at time t - lag
    if (t >= lag)
      acc += cw[d * D_CONV + j] *
             (float)xz[(bt - (size_t)lag) * (2 * D_INNER) + d];
  }
  xconv[i] = (__bf16)silu_f(acc);
}

// ---------------- extract dt_rank slice of x_dbl as bf16 ----------------
__global__ void k_dtr(const float* __restrict__ xdbl, __bf16* __restrict__ dtr) {
  size_t i = (size_t)blockIdx.x * blockDim.x + threadIdx.x;
  if (i >= (size_t)MROWS * DT_RANK) return;
  size_t r = i / DT_RANK;
  int c = (int)(i % DT_RANK);
  dtr[i] = (__bf16)xdbl[r * NPROJ + c];
}

// ---------------- softplus(dt_lin + b_dt), in place ----------------
__global__ void k_softplus(float* __restrict__ dt, const float* __restrict__ bdt) {
  size_t i = (size_t)blockIdx.x * blockDim.x + threadIdx.x;
  if (i >= (size_t)MROWS * D_INNER) return;
  int d = (int)(i % D_INNER);
  float v = dt[i] + bdt[d];
  dt[i] = (v > 20.0f) ? v : log1pf(__expf(v));
}

// ---------------- selective scan, fused gating epilogue ----------------
// One thread per (b, d) channel; 16-state recurrence lives in registers.
// Per-timestep B/C (32 floats, shared by all d) broadcast via LDS.
__global__ __launch_bounds__(256)
void k_scan(const __bf16* __restrict__ xconv, const float* __restrict__ dt,
            const float* __restrict__ xdbl, const __bf16* __restrict__ xz,
            const float* __restrict__ Alog, const float* __restrict__ Dv,
            __bf16* __restrict__ y) {
  __shared__ float sBC[2 * D_STATE];
  const int b = blockIdx.y;
  const int d = blockIdx.x * blockDim.x + threadIdx.x;

  float A[D_STATE], s[D_STATE];
#pragma unroll
  for (int n = 0; n < D_STATE; ++n) {
    A[n] = -__expf(Alog[d * D_STATE + n]);
    s[n] = 0.0f;
  }
  const float Dd = Dv[d];

  for (int t = 0; t < SEQ; ++t) {
    const size_t row = (size_t)b * SEQ + t;
    if (threadIdx.x < 2 * D_STATE)
      sBC[threadIdx.x] = xdbl[row * NPROJ + DT_RANK + threadIdx.x];
    __syncthreads();

    const float xt  = (float)xconv[row * D_INNER + d];
    const float dtt = dt[row * D_INNER + d];
    const float dtx = dtt * xt;
    float yv = 0.0f;
#pragma unroll
    for (int n = 0; n < D_STATE; ++n) {
      float dA = __expf(dtt * A[n]);
      s[n] = s[n] * dA + dtx * sBC[n];
      yv += s[n] * sBC[D_STATE + n];
    }
    const float zv = (float)xz[row * (2 * D_INNER) + D_INNER + d];
    y[row * D_INNER + d] = (__bf16)(yv * silu_f(zv) + Dd * xt);
    __syncthreads();
  }
}

// ---------------------------------------------------------------------------
extern "C" void kernel_launch(void* const* d_in, const int* in_sizes, int n_in,
                              void* d_out, int out_size, void* d_ws, size_t ws_size,
                              hipStream_t stream) {
  (void)in_sizes; (void)n_in; (void)out_size; (void)ws_size;

  const float* x      = (const float*)d_in[0];  // (B,T,D_MODEL)
  const float* W_in   = (const float*)d_in[1];  // (2*D_INNER, D_MODEL)
  const float* conv_w = (const float*)d_in[2];  // (D_INNER,1,D_CONV)
  const float* conv_b = (const float*)d_in[3];  // (D_INNER)
  const float* W_x    = (const float*)d_in[4];  // (NPROJ, D_INNER)
  const float* W_dt   = (const float*)d_in[5];  // (D_INNER, DT_RANK)
  const float* b_dt   = (const float*)d_in[6];  // (D_INNER)
  const float* A_log  = (const float*)d_in[7];  // (D_INNER, D_STATE)
  const float* Dvec   = (const float*)d_in[8];  // (D_INNER)
  const float* W_out  = (const float*)d_in[9];  // (D_MODEL, D_INNER)
  float* out = (float*)d_out;                   // (B,T,D_MODEL) f32

  // ---- workspace carve-out (256B aligned) ----
  char* ws = (char*)d_ws;
  size_t off = 0;
  auto alloc = [&](size_t bytes) -> void* {
    off = (off + 255) & ~(size_t)255;
    void* p = ws + off;
    off += bytes;
    return p;
  };
  __bf16* x_bf    = (__bf16*)alloc((size_t)MROWS * D_MODEL * 2);
  __bf16* Win_bf  = (__bf16*)alloc((size_t)2 * D_INNER * D_MODEL * 2);
  __bf16* Wx_bf   = (__bf16*)alloc((size_t)NPROJ * D_INNER * 2);
  __bf16* Wdt_bf  = (__bf16*)alloc((size_t)D_INNER * DT_RANK * 2);
  __bf16* Wout_bf = (__bf16*)alloc((size_t)D_MODEL * D_INNER * 2);
  __bf16* xz_bf   = (__bf16*)alloc((size_t)MROWS * 2 * D_INNER * 2);
  __bf16* xconv   = (__bf16*)alloc((size_t)MROWS * D_INNER * 2);
  float*  xdbl    = (float*) alloc((size_t)MROWS * NPROJ * 4);
  __bf16* dtr     = (__bf16*)alloc((size_t)MROWS * DT_RANK * 2);
  float*  dtbuf   = (float*) alloc((size_t)MROWS * D_INNER * 4);
  __bf16* ybuf    = (__bf16*)alloc((size_t)MROWS * D_INNER * 2);

  auto cvt = [&](const float* s, __bf16* dst, size_t n) {
    unsigned blocks = (unsigned)((n + 255) / 256);
    k_f32_to_bf16<<<dim3(blocks), dim3(256), 0, stream>>>(s, dst, n);
  };

  // 0) precision staging
  cvt(x,     x_bf,    (size_t)MROWS * D_MODEL);
  cvt(W_in,  Win_bf,  (size_t)2 * D_INNER * D_MODEL);
  cvt(W_x,   Wx_bf,   (size_t)NPROJ * D_INNER);
  cvt(W_dt,  Wdt_bf,  (size_t)D_INNER * DT_RANK);
  cvt(W_out, Wout_bf, (size_t)D_MODEL * D_INNER);

  // 1) in_proj: xz = x @ W_in^T        (16384 x 4096 x 1024) -> bf16
  k_gemm_bf16_wmma<<<dim3((2 * D_INNER) / 128, MROWS / 128), 256, 0, stream>>>(
      x_bf, Win_bf, nullptr, xz_bf, MROWS, 2 * D_INNER, D_MODEL);

  // 2) causal depthwise conv1d + SiLU
  {
    size_t n = (size_t)MROWS * D_INNER;
    k_conv_silu<<<dim3((unsigned)((n + 255) / 256)), 256, 0, stream>>>(
        xz_bf, conv_w, conv_b, xconv);
  }

  // 3) x_proj: x_dbl = x_conv @ W_x^T  (16384 x 96 x 2048) -> f32
  k_gemm_bf16_wmma<<<dim3(1, MROWS / 128), 256, 0, stream>>>(
      xconv, Wx_bf, xdbl, nullptr, MROWS, NPROJ, D_INNER);

  // 4) slice dt_rank columns as bf16
  {
    size_t n = (size_t)MROWS * DT_RANK;
    k_dtr<<<dim3((unsigned)((n + 255) / 256)), 256, 0, stream>>>(xdbl, dtr);
  }

  // 5) dt_proj: dt_lin = dt_r @ W_dt^T (16384 x 2048 x 64) -> f32
  k_gemm_bf16_wmma<<<dim3(D_INNER / 128, MROWS / 128), 256, 0, stream>>>(
      dtr, Wdt_bf, dtbuf, nullptr, MROWS, D_INNER, DT_RANK);

  // 6) dt = softplus(dt_lin + b_dt), in place
  {
    size_t n = (size_t)MROWS * D_INNER;
    k_softplus<<<dim3((unsigned)((n + 255) / 256)), 256, 0, stream>>>(dtbuf, b_dt);
  }

  // 7) selective scan + fused y*silu(z) + D*x_conv -> y (bf16)
  k_scan<<<dim3(D_INNER / 256, BATCH), 256, 0, stream>>>(
      xconv, dtbuf, xdbl, xz_bf, A_log, Dvec, ybuf);

  // 8) out_proj: out = y @ W_out^T     (16384 x 1024 x 2048) -> f32 d_out
  k_gemm_bf16_wmma<<<dim3(D_MODEL / 128, MROWS / 128), 256, 0, stream>>>(
      ybuf, Wout_bf, out, nullptr, MROWS, D_MODEL, D_INNER);
}
